// MultiHeadAttention_59648505807366
// MI455X (gfx1250) — matrix operand order repaired and checked by reading, not compiled, not run on hardware
//
#include <hip/hip_runtime.h>

// MI455X / gfx1250, wave32. All GEMM-shaped work via v_wmma_f32_16x16x32_f16
// (f16 in, f32 accumulate). HBM floor ~7.3us @ 23.3 TB/s for ~170MB traffic;
// ~20 GFLOPs total. Every WMMA operand is a contiguous vector LDS/global load;
// conv D-fragments store transposed so the epilogue is one ds_store_b128.
// proj_kernel stages its A tile with global_load_async_to_lds_b128 (ASYNCcnt).
//
// Fragment layouts (ISA 7.12.2), lane = l, hi = (l>>4)&1:
//   A (16x32 f16): row m = l&15;  element e -> K = hi*16 + e (contiguous order used here)
//   B (32x16 f16): col n = l&15;  element e -> K = hi*16 + e (same order as A)
//   C/D (16x16 f32): reg r -> m = r + hi*8, n = l&15.

typedef __attribute__((ext_vector_type(16))) _Float16 v16h;
typedef __attribute__((ext_vector_type(8)))  _Float16 v8h;
typedef __attribute__((ext_vector_type(4)))  _Float16 v4h;
typedef __attribute__((ext_vector_type(8)))  float    v8f;

#define NB   2048
#define SEQ  10
#define DIM  512
#define PADV 1e-9f

__device__ __forceinline__ v8f wmma_f16(v16h a, v16h b, v8f c) {
  return __builtin_amdgcn_wmma_f32_16x16x32_f16(false, a, false, b, (short)0, c,
                                                false, false);
}
__device__ __forceinline__ v16h cat8(v8h lo, v8h hi8) {
  return __builtin_shufflevector(lo, hi8, 0, 1, 2, 3, 4, 5, 6, 7, 8, 9, 10, 11,
                                 12, 13, 14, 15);
}

// Build one K-half (24 entries) of one im2col column. K is compile-time after
// unrolling, so i=K/3, kk=K%3 fold to constants and the boundary test is a
// single compare (none for kk==1).
template <int LVL, int KLO>
__device__ __forceinline__ void im2col_half(_Float16* __restrict__ cbp,
                                            const _Float16* __restrict__ src,
                                            int c) {
  const int width = 64 << LVL;
  const int head = c >> (6 + LVL);
  const int u = c & (width - 1);
#pragma unroll
  for (int K = KLO; K < KLO + 24; ++K) {
    _Float16 val = (_Float16)0.f;
    if (K < 36) {
      const int i = K / 3, kk = K % 3;
      const int uu = u + kk - 1;  // kernel=3, pad=1
      const bool ok = (kk == 1) || (kk == 0 ? (u > 0) : (u < width - 1));
      if (ok) {
        if (LVL == 0)      val = src[(head * 12 + i) * 64 + uu];
        else if (LVL == 1) val = src[(head * 128 + uu) * 16 + i];
        else               val = src[((head == 0) ? uu : 256 + (uu & 127)) * 16 + i];
      }
    }
    cbp[K] = val;
  }
}

// ---------------------------------------------------------------------------
// Kernel 1: conv pyramid for one (tensor, batch). Each conv level is an
// im2col GEMM: out[o][c] = sum_{K<36} Wcol[o][K] * col[c][K].
// col is materialized in LDS 128 columns/pass, K padded to 48 (zeros) so
// B-fragments are single 32B ds loads. Conv outputs for levels 1/2 are kept
// transposed in LDS ([f][16 rows]) so the D epilogue is one ds_store_b128;
// rows <10 are simultaneously streamed to global (the attention layout).
// Level 3 output goes straight to global (rows 10/11 are never needed).
// ---------------------------------------------------------------------------
__global__ __launch_bounds__(256) void pyramid_kernel(
    const float* __restrict__ q, const float* __restrict__ k,
    const float* __restrict__ v, const float* __restrict__ convw,
    const float* __restrict__ convb, _Float16* __restrict__ pyr) {
  // [0,16384): phase1 x_f16 (10KB) -> phase2 Wcol(2KB)@0 + colbuf(12KB)@4096, overlaps c1_in
  // [14336,26624): c1_in [8][12][64]   12KB
  // [26624,43008): c1^T  [512][16]     16KB
  // [43008,59392): c2^T  [512][16]     16KB
  __shared__ __align__(32) unsigned char smem[59392];
  _Float16* xsh    = (_Float16*)smem;             // [10][512] f16
  _Float16* Wcol   = (_Float16*)smem;             // [16][64]
  _Float16* colbuf = (_Float16*)(smem + 4096);    // [128][48]
  _Float16* c1s    = (_Float16*)(smem + 14336);   // [8][12][64]
  _Float16* cAT    = (_Float16*)(smem + 26624);   // [512][16]
  _Float16* cBT    = (_Float16*)(smem + 43008);   // [512][16]

  const int b = blockIdx.x, t = blockIdx.y;
  const int tid = threadIdx.x, lane = tid & 31, wave = tid >> 5;
  const int hi = (lane >> 4) & 1;
  const float* xg = (t == 0 ? q : (t == 1 ? k : v)) + (size_t)b * SEQ * DIM;

  for (int idx = tid; idx < (SEQ * DIM) / 4; idx += 256) {
    float4 xv = ((const float4*)xg)[idx];
    v4h hv = {(_Float16)xv.x, (_Float16)xv.y, (_Float16)xv.z, (_Float16)xv.w};
    *(v4h*)(xsh + idx * 4) = hv;
  }
  __syncthreads();

  // c1_in[h][i][t] = sum over neighbor heads (zero-pad in h); rows i>=10 use
  // PAD_VAL per neighbor.
  for (int idx = tid; idx < 8 * 12 * 64; idx += 256) {
    int tt = idx & 63, i = (idx >> 6) % 12, h = idx / (12 * 64);
    float acc = 0.f;
#pragma unroll
    for (int dh = -1; dh <= 1; ++dh) {
      int hh = h + dh;
      if (hh >= 0 && hh < 8)
        acc += (i < SEQ) ? (float)xsh[i * DIM + hh * 64 + tt] : PADV;
    }
    c1s[idx] = (_Float16)acc;
  }
  __syncthreads();  // x region now reusable (Wcol + colbuf)

  // Wcol[o][K] (K padded to 64 with zeros, rows o>=12 zero), K = i*3 + kk
  for (int idx = tid; idx < 1024; idx += 256) {
    int o = idx >> 6, kk = idx & 63;
    float w = (kk < 36 && o < 12) ? convw[(o * 12 + kk / 3) * 3 + (kk % 3)] : 0.f;
    Wcol[idx] = (_Float16)w;
  }
  __syncthreads();

  // A fragments (weights) + per-lane bias registers, for all levels
  const int o = lane & 15;
  v16h aw[2];
#pragma unroll
  for (int st = 0; st < 2; ++st) {
    v8h a0 = *(const v8h*)(Wcol + o * 64 + st * 32 + hi * 16);
    v8h a1 = *(const v8h*)(Wcol + o * 64 + st * 32 + hi * 16 + 8);
    aw[st] = cat8(a0, a1);
  }
  float bb[8];
#pragma unroll
  for (int r = 0; r < 8; ++r) {
    int oo = r + hi * 8;
    bb[r] = (oo < 12) ? convb[oo] : 0.f;
  }

#pragma unroll
  for (int lvl = 0; lvl < 3; ++lvl) {
    _Float16* dstT = (lvl == 0) ? cAT : cBT;  // unused when lvl==2
    const size_t ob = ((size_t)(t * 3 + lvl) * NB + b) * 5120;
#pragma unroll 1
    for (int p = 0; p < 4; ++p) {  // 4 passes x 128 columns (keep rolled)
      {  // one (column, K-half) per thread; wave-uniform split
        const int cl = tid & 127;
        const int c0 = p * 128 + cl;
        _Float16* cbp = colbuf + cl * 48;
        if (tid < 128) {
          if (lvl == 0)      im2col_half<0, 0>(cbp, c1s, c0);
          else if (lvl == 1) im2col_half<1, 0>(cbp, cAT, c0);
          else               im2col_half<2, 0>(cbp, cBT, c0);
        } else {
          if (lvl == 0)      im2col_half<0, 24>(cbp, c1s, c0);
          else if (lvl == 1) im2col_half<1, 24>(cbp, cAT, c0);
          else               im2col_half<2, 24>(cbp, cBT, c0);
        }
      }
      __syncthreads();

      // one 16-column tile per wave
      const int cl = wave * 16 + (lane & 15);
      const int c = p * 128 + cl;
      v16h b0 = *(const v16h*)(colbuf + cl * 48 + hi * 16);  // abs K 0..31
      v16h b1 = {};                                          // abs K 32..63
      if (hi == 0) b1 = *(const v16h*)(colbuf + cl * 48 + 32);
      v8f acc = {};
      acc = wmma_f16(aw[0], b0, acc);
      acc = wmma_f16(aw[1], b1, acc);

      v8h hv;
#pragma unroll
      for (int r = 0; r < 8; ++r) hv[r] = (_Float16)(acc[r] + bb[r]);
      if (lvl < 2) *(v8h*)(dstT + c * 16 + hi * 8) = hv;  // rows hi*8..hi*8+7

      // stream rows <10 to global: low half stores rows 0..7, high half 8..9
      if (hi == 0) {
#pragma unroll
        for (int r = 0; r < 8; ++r) pyr[ob + (size_t)r * DIM + c] = hv[r];
      } else {
        pyr[ob + (size_t)8 * DIM + c] = hv[0];
        pyr[ob + (size_t)9 * DIM + c] = hv[1];
      }
      __syncthreads();
    }
  }
}

// ---------------------------------------------------------------------------
// Kernel 2: 3-level attention for one batch; wave = head. Q/K fragments are
// unguarded contiguous global vector loads (over-reads hit adjacent finite
// workspace); garbage rows/cols are killed by the explicitly zero-padded P.
// V is transposed per level into LDS so PV B-fragments are single 32B ds
// loads. xacc has a shared dump row (16) so C-frag LDS traffic is unguarded.
// ---------------------------------------------------------------------------
__global__ __launch_bounds__(256) void attn_kernel(
    const _Float16* __restrict__ pyr, const int* __restrict__ mask,
    _Float16* __restrict__ xout) {
  __shared__ __align__(32) float    xacc[17 * DIM];  // rows 0..9 + dump row 16
  __shared__ __align__(32) _Float16 vT[DIM * 16];    // v^T, zero-padded
  __shared__ __align__(32) unsigned char wsc[8][1024];  // per-wave sc/P region

  const int b = blockIdx.x;
  const int tid = threadIdx.x, lane = tid & 31, wave = tid >> 5;
  const int hi = (lane >> 4) & 1;
  float*    scw = (float*)&wsc[wave][0];     // 16x16 f32 scores
  _Float16* pw  = (_Float16*)&wsc[wave][0];  // then 16x32 f16 probs

  for (int i = tid; i < 17 * DIM; i += 256) xacc[i] = 0.f;
  __syncthreads();

  for (int lvl = 0; lvl < 3; ++lvl) {
    const int d = 64 << lvl, heads = 8 >> lvl;
    const _Float16* vb = pyr + ((size_t)(2 * 3 + lvl) * NB + b) * 5120;

    // vT[f][kk] = v[kk][f] for kk<10 else 0
    for (int ch = tid; ch < 1024; ch += 256) {
      int kk = ch >> 6, f0 = (ch & 63) * 8;
      v8h vv = {};
      if (kk < SEQ) vv = *(const v8h*)(vb + kk * DIM + f0);
#pragma unroll
      for (int j = 0; j < 8; ++j) vT[(f0 + j) * 16 + kk] = vv[j];
    }
    __syncthreads();

    if (wave < heads) {  // wave-uniform => EXEC all-ones at every WMMA
      const int h = wave;
      const _Float16* qb = pyr + ((size_t)(0 * 3 + lvl) * NB + b) * 5120;
      const _Float16* kb = pyr + ((size_t)(1 * 3 + lvl) * NB + b) * 5120;

      // scores = q k^T  (M=N=16 from 10, K=d)
      v8f cs = {};
      for (int ks = 0; ks < d; ks += 32) {
        const _Float16* qp = qb + (lane & 15) * DIM + h * d + ks;
        v16h af = cat8(*(const v8h*)(qp + hi * 16),
                       *(const v8h*)(qp + hi * 16 + 8));
        v16h bf = *(const v16h*)(kb + (lane & 15) * DIM + h * d + ks + hi * 16);
        cs = wmma_f16(af, bf, cs);
      }
#pragma unroll
      for (int r = 0; r < 8; ++r)
        scw[(r + hi * 8) * 16 + (lane & 15)] = cs[r];

      // masked scaled softmax, one row per lane (lanes 0..9); the score
      // region is fully consumed into registers before pw overwrites it
      // (LDS ops from one wave stay in order).
      float srow[SEQ], inv = 0.f;
      if (lane < SEQ) {
        const float scale = rsqrtf((float)d);
        float mx = -1e30f;
#pragma unroll
        for (int j = 0; j < SEQ; ++j) {
          float sv = scw[lane * 16 + j] * scale;
          if (mask[(size_t)b * SEQ * SEQ + lane * SEQ + j] == 0) sv = -1e9f;
          srow[j] = sv;
          mx = fmaxf(mx, sv);
        }
        float sum = 0.f;
#pragma unroll
        for (int j = 0; j < SEQ; ++j) {
          srow[j] = __expf(srow[j] - mx);
          sum += srow[j];
        }
        inv = 1.f / sum;
      }
      v16h z16 = {};
      *(v16h*)(pw + lane * 16) = z16;  // zero whole P region (32B per lane)
      if (lane < SEQ) {
#pragma unroll
        for (int j = 0; j < SEQ; ++j)
          pw[lane * 32 + j] = (_Float16)(srow[j] * inv);
      }

      // x[h-slice] += P @ V   (K=32 from 10, N=d)
      const _Float16* pp = pw + (lane & 15) * 32;
      v16h ap = cat8(*(const v8h*)(pp + hi * 16), *(const v8h*)(pp + hi * 16 + 8));
      for (int nt = 0; nt < (d >> 4); ++nt) {
        const int f = h * d + nt * 16 + (lane & 15);
        v16h bv = {};  // K rows 16..31 are zero (only 10 keys)
        if (hi == 0) bv = *(const v16h*)(vT + f * 16);
        v8f cx;
#pragma unroll
        for (int r = 0; r < 8; ++r) {
          int m = r + hi * 8;
          int mr = (m < SEQ) ? m : 16;  // dump row: benign cross-wave races
          cx[r] = xacc[mr * DIM + f];
        }
        cx = wmma_f16(ap, bv, cx);
#pragma unroll
        for (int r = 0; r < 8; ++r) {
          int m = r + hi * 8;
          int mr = (m < SEQ) ? m : 16;
          xacc[mr * DIM + f] = cx[r];
        }
      }
    }
    __syncthreads();
  }

  for (int ch = tid; ch < 640; ch += 256) {
    int off = ch * 8;
    v8h ov;
#pragma unroll
    for (int j = 0; j < 8; ++j) ov[j] = (_Float16)xacc[off + j];
    *(v8h*)(xout + (size_t)b * 5120 + off) = ov;
  }
}

// ---------------------------------------------------------------------------
// Kernel 3: out = x @ w_out^T + b_out.  (20480 x 512) x (512 x 512).
// WG tile 64M x 256N; wave tile 16M x 128N (8 accumulators). W tile staged as
// [n][k] so B-fragments are single 32B ds loads. The A tile is staged with
// the gfx1250 async global->LDS DMA path (ASYNCcnt).
// ---------------------------------------------------------------------------
__global__ __launch_bounds__(256) void proj_kernel(
    const _Float16* __restrict__ x, const float* __restrict__ wgt,
    const float* __restrict__ bias, float* __restrict__ out) {
  __shared__ __align__(32) _Float16 xs[64 * 32];    // A tile [m][k]  4KB
  __shared__ __align__(32) _Float16 wst[256 * 32];  // B tile [n][k] 16KB

  const int tid = threadIdx.x, lane = tid & 31, wave = tid >> 5;
  const int hi = (lane >> 4) & 1;
  const int mbase = blockIdx.x * 64;
  const int nbase = blockIdx.y * 256;
  const int mt = wave >> 1;  // 16-row tile
  const int nh = wave & 1;   // 128-col strip

  v8f acc[8] = {};

#pragma unroll 1
  for (int k0 = 0; k0 < DIM; k0 += 32) {
    {  // stage A tile: one async 16B global->LDS copy per thread
      int r = tid >> 2, c8 = (tid & 3) * 8;
      const _Float16* gp = x + (size_t)(mbase + r) * DIM + k0 + c8;
      unsigned lds = (unsigned)(size_t)(xs + r * 32 + c8);  // aperture low bits
      asm volatile("global_load_async_to_lds_b128 %0, %1, off"
                   :: "v"(lds), "v"(gp) : "memory");
    }
    for (int ch = tid; ch < 1024; ch += 256) {  // stage B tile (f32 -> f16)
      int n = ch >> 2, k8 = (ch & 3) * 8;
      const float* wp = wgt + (size_t)(nbase + n) * DIM + k0 + k8;
      v8h hv;
#pragma unroll
      for (int j = 0; j < 8; ++j) hv[j] = (_Float16)wp[j];
      *(v8h*)(wst + n * 32 + k8) = hv;
    }
    asm volatile("s_wait_asynccnt 0x0" ::: "memory");  // drain async DMA
    __syncthreads();

    const _Float16* xp = xs + (mt * 16 + (lane & 15)) * 32;
    v16h af = cat8(*(const v8h*)(xp + hi * 16), *(const v8h*)(xp + hi * 16 + 8));
#pragma unroll
    for (int nt = 0; nt < 8; ++nt) {
      int n = nh * 128 + nt * 16 + (lane & 15);
      v16h bf = *(const v16h*)(wst + n * 32 + hi * 16);
      acc[nt] = wmma_f16(af, bf, acc[nt]);
    }
    __syncthreads();
  }

#pragma unroll
  for (int nt = 0; nt < 8; ++nt) {
#pragma unroll
    for (int r = 0; r < 8; ++r) {
      int m = mbase + mt * 16 + r + hi * 8;
      int n = nbase + nh * 128 + nt * 16 + (lane & 15);
      out[(size_t)m * DIM + n] = acc[nt][r] + bias[n];
    }
  }
}

// ---------------------------------------------------------------------------
// Host wrapper. Inputs (setup_inputs order): query, key, value, mask, conv_w,
// conv_b, w_out, b_out. Output: fp32 (2048,10,512).
// Workspace (f16): pyr[9][NB][5120] then xbuf[NB][5120]  (~200 MB).
// ---------------------------------------------------------------------------
extern "C" void kernel_launch(void* const* d_in, const int* in_sizes, int n_in,
                              void* d_out, int out_size, void* d_ws,
                              size_t ws_size, hipStream_t stream) {
  const float* q      = (const float*)d_in[0];
  const float* k      = (const float*)d_in[1];
  const float* v      = (const float*)d_in[2];
  const int*   mask   = (const int*)d_in[3];
  const float* conv_w = (const float*)d_in[4];
  const float* conv_b = (const float*)d_in[5];
  const float* w_out  = (const float*)d_in[6];
  const float* b_out  = (const float*)d_in[7];
  float* out = (float*)d_out;

  _Float16* pyr = (_Float16*)d_ws;              // 9 * NB * 5120 halves
  _Float16* xb  = pyr + (size_t)9 * NB * 5120;  // NB * 5120 halves

  pyramid_kernel<<<dim3(NB, 3), 256, 0, stream>>>(q, k, v, conv_w, conv_b, pyr);
  attn_kernel<<<dim3(NB), 256, 0, stream>>>(pyr, mask, xb);
  proj_kernel<<<dim3((NB * SEQ) / 64, DIM / 256), 256, 0, stream>>>(xb, w_out,
                                                                    b_out, out);
}
